// VectorQuantiser_28355374088288
// MI455X (gfx1250) — matrix operand order repaired and checked by reading, not compiled, not run on hardware
//
#include <hip/hip_runtime.h>

typedef __attribute__((ext_vector_type(16))) __bf16 v16bf;
typedef __attribute__((ext_vector_type(8)))  float  v8f;
typedef __attribute__((ext_vector_type(4)))  float  f4;

#define B_      32
#define D_      64
#define L_      4096
#define NEMB    1024
#define ROWS    (B_ * L_)        // 131072
#define ZQ_ELEMS ((size_t)ROWS * D_)  // 8388608
#define LOSS_OFF ZQ_ELEMS
#define CODES_OFF (ZQ_ELEMS + 1)

// d_ws layout: floats [0,1024) = ||e||^2 ; floats [1024,2048) = per-WG loss partials
// bytes 8192+ : packed bf16 codebook fragments (65536 bf16 = 128 KB)

// dynamic LDS layout for vq_main (bytes):
//   embp   : [0, 131072)        bf16 packed B fragments
//   zt     : [131072, 163840)   float z tile, zt[d*128 + j]
//   esq    : [163840, 167936)   float ||e||^2
//   codesL : [167936, 168448)   int codes for the 128 rows
//   red    : [168448, 169472)   float block-reduce scratch
#define LDS_BYTES 169472

// ---------------- prep: ||e||^2 + bf16 fragment-packed codebook ----------------
__global__ void vq_prep(const float* __restrict__ emb,
                        float* __restrict__ wsF,
                        __bf16* __restrict__ wsPack) {
  int gid = blockIdx.x * 256 + threadIdx.x;
  if (gid < NEMB) {
    const float* e = emb + gid * D_;
    float s = 0.f;
    #pragma unroll
    for (int k = 0; k < D_; ++k) { float v = e[k]; s += v * v; }
    wsF[gid] = s;
  }
  // packed element: flat = ((t*2+h)*32 + lane)*16 + e
  // B fragment layout (16-bit B 32x16): lane = N%16 (+16 lanes hold K upper half),
  // element e -> K = h*32 + (laneHi ? 8 : 0) + (e < 8 ? e : e + 8)
  if (gid < 65536) {
    int e  = gid & 15;
    int l  = (gid >> 4) & 31;
    int th = gid >> 9;          // t*2 + h
    int h  = th & 1;
    int t  = th >> 1;
    int n  = t * 16 + (l & 15);
    int k  = h * 32 + ((l >= 16) ? 8 : 0) + ((e < 8) ? e : e + 8);
    wsPack[gid] = (__bf16)emb[n * D_ + k];
  }
}

// ---------------- main fused kernel ----------------
__global__ void vq_main(const float* __restrict__ z,
                        const float* __restrict__ emb,
                        const float* __restrict__ wsF,
                        const __bf16* __restrict__ wsPack,
                        float* __restrict__ out,
                        float* __restrict__ partial) {
  extern __shared__ char smem[];
  __bf16* embp   = (__bf16*)smem;
  float*  zt     = (float*)(smem + 131072);
  float*  esq    = (float*)(smem + 163840);
  int*    codesL = (int*)(smem + 167936);
  float*  red    = (float*)(smem + 168448);

  const int tid  = threadIdx.x;
  const int wg   = blockIdx.x;
  const int row0 = wg * 128;          // always within one batch index b
  const int b    = row0 >> 12;
  const int l0   = row0 & 4095;

  // packed codebook (128 KB) global -> LDS, 16B vectors
  {
    const f4* src = (const f4*)wsPack;
    f4* dst = (f4*)embp;
    #pragma unroll
    for (int i = 0; i < 32; ++i) dst[tid + i * 256] = src[tid + i * 256];
  }
  // ||e||^2 -> LDS
  #pragma unroll
  for (int i = 0; i < 4; ++i) esq[tid + i * 256] = wsF[tid + i * 256];
  // z tile -> LDS (transpose staging): zt[d*128 + j] = z[(b*64+d)*4096 + l0 + j]
  #pragma unroll
  for (int i = 0; i < 8; ++i) {
    int idx4 = tid + i * 256;         // 0..2047 float4s
    int d  = idx4 >> 5;
    int j4 = idx4 & 31;
    f4 v = *(const f4*)(z + (size_t)(b * D_ + d) * L_ + l0 + j4 * 4);
    *(f4*)(zt + d * 128 + j4 * 4) = v;
  }
  __syncthreads();

  const int lane   = tid & 31;
  const int waveId = tid >> 5;
  const int col    = lane & 15;
  const int kq     = (lane < 16) ? 0 : 8;   // A-matrix K offset for hi lanes

  // Build the two bf16 A fragments (16-bit A 16x32 layout, reused for all 64 N tiles)
  v16bf A0, A1;
  #pragma unroll
  for (int e = 0; e < 16; ++e) {
    int kk = kq + ((e < 8) ? e : e + 8);
    A0[e] = (__bf16)zt[kk * 128 + waveId * 16 + col];
    A1[e] = (__bf16)zt[(kk + 32) * 128 + waveId * 16 + col];
  }

  float best[8]; int bidx[8];
  #pragma unroll
  for (int v = 0; v < 8; ++v) { best[v] = 3.4e38f; bidx[v] = 0; }

  // ---- software-pipelined GEMM + argmin: issue WMMAs for tile t, then run the
  // ---- epilogue of tile t-1 (puts ~30 VALU ops between a WMMA and the VALU
  // ---- that reads its D-matrix -> no hazard nops, XDL/VALU co-execution)
  v8f  accP;
  int  nP;
  float esP;
  {
    const v16bf B0 = *(const v16bf*)(embp + (0 * 32 + lane) * 16);
    const v16bf B1 = *(const v16bf*)(embp + (1 * 32 + lane) * 16);
    v8f acc = {};
    acc = __builtin_amdgcn_wmma_f32_16x16x32_bf16(false, A0, false, B0,
                                                  (short)0, acc, false, false);
    acc = __builtin_amdgcn_wmma_f32_16x16x32_bf16(false, A1, false, B1,
                                                  (short)0, acc, false, false);
    accP = acc; nP = col; esP = esq[col];
  }
  for (int t = 1; t < 64; ++t) {
    const v16bf B0 = *(const v16bf*)(embp + ((t * 2 + 0) * 32 + lane) * 16);
    const v16bf B1 = *(const v16bf*)(embp + ((t * 2 + 1) * 32 + lane) * 16);
    v8f acc = {};
    acc = __builtin_amdgcn_wmma_f32_16x16x32_bf16(false, A0, false, B0,
                                                  (short)0, acc, false, false);
    acc = __builtin_amdgcn_wmma_f32_16x16x32_bf16(false, A1, false, B1,
                                                  (short)0, acc, false, false);
    const int   n  = t * 16 + col;
    const float es = esq[n];
    // epilogue for previous tile (independent of this tile's WMMAs)
    #pragma unroll
    for (int v = 0; v < 8; ++v) {
      float dist = esP - 2.0f * accP[v];   // ||z||^2 constant per row: dropped
      if (dist < best[v]) { best[v] = dist; bidx[v] = nP; }
    }
    accP = acc; nP = n; esP = es;
  }
  #pragma unroll
  for (int v = 0; v < 8; ++v) {
    float dist = esP - 2.0f * accP[v];
    if (dist < best[v]) { best[v] = dist; bidx[v] = nP; }
  }

  // argmin butterfly within each 16-lane group (lanes 0-15: M=v, 16-31: M=v+8)
  #pragma unroll
  for (int v = 0; v < 8; ++v) {
    float bd = best[v]; int bi = bidx[v];
    #pragma unroll
    for (int m = 8; m >= 1; m >>= 1) {
      float od = __shfl_xor(bd, m, 32);
      int   oi = __shfl_xor(bi, m, 32);
      if (od < bd || (od == bd && oi < bi)) { bd = od; bi = oi; }
    }
    if (lane == 0) {
      codesL[waveId * 16 + v] = bi;
      out[CODES_OFF + row0 + waveId * 16 + v] = (float)bi;
    }
    if (lane == 16) {
      codesL[waveId * 16 + v + 8] = bi;
      out[CODES_OFF + row0 + waveId * 16 + v + 8] = (float)bi;
    }
  }
  __syncthreads();

  // z_q gather (codebook is L2-hot) + coalesced store along l + fused loss
  float lsum = 0.f;
  #pragma unroll
  for (int i = 0; i < 32; ++i) {
    int flat = tid + i * 256;   // 0..8191
    int d = flat >> 7;
    int j = flat & 127;
    int c = codesL[j];
    float q = emb[c * D_ + d];
    float diff = zt[d * 128 + j] - q;
    lsum += diff * diff;
    out[(size_t)(b * D_ + d) * L_ + l0 + j] = q;
  }
  red[tid] = lsum;
  __syncthreads();
  for (int s = 128; s > 0; s >>= 1) {
    if (tid < s) red[tid] += red[tid + s];
    __syncthreads();
  }
  if (tid == 0) partial[wg] = red[0];
}

// ---------------- deterministic loss reduction ----------------
__global__ void vq_finish(const float* __restrict__ partial, float* __restrict__ out) {
  __shared__ float red[256];
  int tid = threadIdx.x;
  float s = 0.f;
  #pragma unroll
  for (int i = 0; i < 4; ++i) s += partial[tid + i * 256];
  red[tid] = s;
  __syncthreads();
  for (int k = 128; k > 0; k >>= 1) {
    if (tid < k) red[tid] += red[tid + k];
    __syncthreads();
  }
  // loss = vq_loss + 0.25*commitment = 1.25 * mean((z - z_q)^2)
  if (tid == 0) out[LOSS_OFF] = 1.25f * red[0] / (float)ZQ_ELEMS;
}

extern "C" void kernel_launch(void* const* d_in, const int* in_sizes, int n_in,
                              void* d_out, int out_size, void* d_ws, size_t ws_size,
                              hipStream_t stream) {
  (void)in_sizes; (void)n_in; (void)out_size; (void)ws_size;
  const float* z   = (const float*)d_in[0];
  const float* emb = (const float*)d_in[1];
  float*  out     = (float*)d_out;
  float*  wsF     = (float*)d_ws;
  __bf16* wsPack  = (__bf16*)((char*)d_ws + 8192);
  float*  partial = wsF + 1024;

  hipFuncSetAttribute((const void*)vq_main,
                      hipFuncAttributeMaxDynamicSharedMemorySize, LDS_BYTES);

  vq_prep<<<256, 256, 0, stream>>>(emb, wsF, wsPack);
  vq_main<<<1024, 256, LDS_BYTES, stream>>>(z, emb, wsF, wsPack, out, partial);
  vq_finish<<<1, 256, 0, stream>>>(partial, out);
}
